// Separator_55276229099688
// MI455X (gfx1250) — matrix operand order repaired and checked by reading, not compiled, not run on hardware
//
#include <hip/hip_runtime.h>

// ---------------------------------------------------------------------------
// Conv-TasNet separator for MI455X (gfx1250), wave32 + WMMA f16.
// All GroupNorms are folded into the following 1x1 GEMM epilogue:
//   W @ (rstd*g*(x-mean) + b) = rstd*(W.g) @ x + (u - mean*rstd*v)
// so each TCN block is only 2 kernels (dwconv+prelu+stats, WMMA gemm+residual).
// Deterministic: reductions use fixed partial slots + finalize kernel.
// Weight tiles staged to LDS via async global->LDS copies when available.
// ---------------------------------------------------------------------------

typedef __attribute__((ext_vector_type(16))) _Float16 h16;
typedef __attribute__((ext_vector_type(8)))  _Float16 h8;
typedef __attribute__((ext_vector_type(8)))  float    f8;
typedef __attribute__((ext_vector_type(4)))  int      i4;

#define N_CH   512
#define B_CH   128
#define NBLK   24
#define NSPK   2
#define LLEN   8000
#define SAMPLES 4
#define EPS_GN 1e-5f

#define EPI_BIAS 0   // y = rstd*acc + (u - mean*rstd*v)
#define EPI_RES  1   // y = rstd*acc + bias + dst_old   (in-place residual)
#define EPI_SIG  2   // y = sigmoid(acc)

#if __has_builtin(__builtin_amdgcn_global_load_async_to_lds_b128) && \
    __has_builtin(__builtin_amdgcn_s_wait_asynccnt)
#define USE_ASYNC_LDS 1
typedef __attribute__((address_space(1))) i4* gp1_t;   // global int4*
typedef __attribute__((address_space(3))) i4* lp3_t;   // LDS int4*
#else
#define USE_ASYNC_LDS 0
#endif

// ---------------------------------------------------------------------------
// Weight prep: fold per-channel gamma into f16 weights, copy w_exp to f16.
// ---------------------------------------------------------------------------
__global__ void prep_weights_kernel(const float* __restrict__ w_comp,
                                    const float* __restrict__ gn_g,
                                    const float* __restrict__ pw_w,
                                    const float* __restrict__ tn_g,
                                    const float* __restrict__ w_exp,
                                    _Float16* __restrict__ Wg,
                                    _Float16* __restrict__ PWg,
                                    _Float16* __restrict__ Wexp) {
  int idx = blockIdx.x * blockDim.x + threadIdx.x;
  const int t1 = B_CH * N_CH;            // 65536
  const int t2 = NBLK * B_CH * B_CH;     // 393216
  const int t3 = NSPK * N_CH * B_CH;     // 131072
  if (idx < t1) {
    int n = idx % N_CH;
    Wg[idx] = (_Float16)(w_comp[idx] * gn_g[n]);
  } else if (idx < t1 + t2) {
    int rel = idx - t1;
    int i = rel / (B_CH * B_CH);
    int c = rel % B_CH;
    PWg[rel] = (_Float16)(pw_w[rel] * tn_g[i * B_CH + c]);
  } else if (idx < t1 + t2 + t3) {
    int rel = idx - t1 - t2;
    Wexp[rel] = (_Float16)w_exp[rel];
  }
}

// u[r] = sum_k W[r,k]*beta[k],  v[r] = sum_k W[r,k]*gamma[k]
__global__ void prep_uv_kernel(const float* __restrict__ w_comp,
                               const float* __restrict__ gn_g,
                               const float* __restrict__ gn_b,
                               const float* __restrict__ pw_w,
                               const float* __restrict__ tn_g,
                               const float* __restrict__ tn_b,
                               float* __restrict__ u, float* __restrict__ v) {
  int r = blockIdx.x * blockDim.x + threadIdx.x;
  if (r >= (1 + NBLK) * B_CH) return;
  float su = 0.f, sv = 0.f;
  if (r < B_CH) {
    const float* wr = w_comp + (long)r * N_CH;
    for (int n = 0; n < N_CH; ++n) { su += wr[n] * gn_b[n]; sv += wr[n] * gn_g[n]; }
  } else {
    int rr = r - B_CH;
    int i = rr / B_CH, o = rr % B_CH;
    const float* wr = pw_w + ((long)i * B_CH + o) * B_CH;
    const float* tg = tn_g + i * B_CH;
    const float* tb = tn_b + i * B_CH;
    for (int c = 0; c < B_CH; ++c) { su += wr[c] * tb[c]; sv += wr[c] * tg[c]; }
  }
  u[r] = su; v[r] = sv;
}

// ---------------------------------------------------------------------------
// Reductions (deterministic: fixed partial slots, then finalize).
// ---------------------------------------------------------------------------
__device__ __forceinline__ void block_reduce_2(float sum, float ssq,
                                               float* out0, float* out1) {
  __shared__ float sh0[256], sh1[256];
  int tid = threadIdx.x;
  sh0[tid] = sum; sh1[tid] = ssq;
  __syncthreads();
  for (int st = 128; st > 0; st >>= 1) {
    if (tid < st) { sh0[tid] += sh0[tid + st]; sh1[tid] += sh1[tid + st]; }
    __syncthreads();
  }
  if (tid == 0) { *out0 = sh0[0]; *out1 = sh1[0]; }
}

// Per-sample sum/sumsq of x [S][N_CH][L] -> partials [S][128][2]
__global__ void reduce_x_kernel(const float* __restrict__ x,
                                float* __restrict__ partials) {
  const int s = blockIdx.y, b = blockIdx.x;            // 128 blocks per sample
  const long per = (long)N_CH * LLEN;
  float sum = 0.f, ssq = 0.f;
  for (long i = (long)b * 256 + threadIdx.x; i < per; i += 128L * 256) {
    float vv = x[(long)s * per + i];
    sum += vv; ssq += vv * vv;
  }
  long pidx = ((long)s * 128 + b) * 2;
  block_reduce_2(sum, ssq, &partials[pidx], &partials[pidx + 1]);
}

__global__ void finalize_stats_kernel(const float* __restrict__ partials,
                                      int nslots, float* __restrict__ stats) {
  const int s = blockIdx.x;
  float sum = 0.f, ssq = 0.f;
  for (int i = threadIdx.x; i < nslots; i += 256) {
    sum += partials[((long)s * nslots + i) * 2];
    ssq += partials[((long)s * nslots + i) * 2 + 1];
  }
  block_reduce_2(sum, ssq, &stats[s * 2], &stats[s * 2 + 1]);
}

// ---------------------------------------------------------------------------
// Depthwise causal dilated conv (K=3) + PReLU + f16 store + stats partials.
// grid (32, 128, 4): (L/256, channels, samples)
// ---------------------------------------------------------------------------
__global__ void dwprelu_kernel(const float* __restrict__ outbuf,
                               const float* __restrict__ dw_w,
                               const float* __restrict__ pr_a,
                               _Float16* __restrict__ tbuf,
                               float* __restrict__ partials,
                               int blk, int d) {
  const int s = blockIdx.z, c = blockIdx.y;
  const int l = blockIdx.x * 256 + threadIdx.x;
  const float w0 = dw_w[(blk * B_CH + c) * 3 + 0];
  const float w1 = dw_w[(blk * B_CH + c) * 3 + 1];
  const float w2 = dw_w[(blk * B_CH + c) * 3 + 2];
  const float a  = pr_a[blk * B_CH + c];
  const long base = ((long)s * B_CH + c) * LLEN;
  float sum = 0.f, ssq = 0.f;
  if (l < LLEN) {
    float v2 = outbuf[base + l];
    float v1 = (l >= d)     ? outbuf[base + l - d]     : 0.f;
    float v0 = (l >= 2 * d) ? outbuf[base + l - 2 * d] : 0.f;
    float val = w0 * v0 + w1 * v1 + w2 * v2;
    val = val > 0.f ? val : a * val;      // PReLU
    tbuf[base + l] = (_Float16)val;
    sum = val; ssq = val * val;
  }
  long pidx = (((long)s * (B_CH * 32)) + c * 32 + blockIdx.x) * 2;
  block_reduce_2(sum, ssq, &partials[pidx], &partials[pidx + 1]);
}

// ---------------------------------------------------------------------------
// WMMA GEMM: dst[s][M][L] (+epilogue) = W_f16[M][K] @ src[s][K][L]
// Block = 256 thr (8 waves), 128-col L tile. Wave w owns col-tile w, loops
// 4 row-tiles of the current 64-row M chunk.
// ---------------------------------------------------------------------------
template <int M, int K, int EPI, bool SRCF16>
__global__ void gemm_kernel(const _Float16* __restrict__ Wf16,
                            const void* __restrict__ src,
                            float* __restrict__ dst,
                            const float* __restrict__ stats,
                            const float* __restrict__ u,
                            const float* __restrict__ v,
                            float invCnt) {
  constexpr int XK = 128;        // K chunk
  constexpr int XS = XK + 8;     // padded LDS row stride (halfs), keeps 16B align
  constexpr int KT = K / XK;     // K chunks
  constexpr int MH = M / 64;     // 64-row M chunks
  __shared__ __attribute__((aligned(32))) _Float16 Wlds[64 * XK];
  __shared__ __attribute__((aligned(32))) _Float16 Xlds[128 * XS];

  const int s    = blockIdx.y;
  const int l0   = blockIdx.x * 128;
  const int tid  = threadIdx.x;
  const int lane = tid & 31;
  const int wave = tid >> 5;          // col tile 0..7
  const float* srcF = (const float*)src;
  const _Float16* srcH = (const _Float16*)src;

  float rstd = 1.f, mean_rstd = 0.f;
  if (EPI != EPI_SIG) {
    float sum = stats[s * 2], ssq = stats[s * 2 + 1];
    float mean = sum * invCnt;
    float var  = ssq * invCnt - mean * mean;
    rstd = rsqrtf(var + EPS_GN);
    mean_rstd = mean * rstd;
  }

  for (int mh = 0; mh < MH; ++mh) {
    f8 acc[4];
#pragma unroll
    for (int rt = 0; rt < 4; ++rt)
#pragma unroll
      for (int j = 0; j < 8; ++j) acc[rt][j] = 0.f;

    for (int kc = 0; kc < KT; ++kc) {
      // ---- stage weight tile row-major: Wlds[r][k] (64x128 halfs)
      {
        const int r  = tid >> 2;
        const int c0 = (tid & 3) * 32;
        const _Float16* gsrc = Wf16 + ((long)(mh * 64 + r) * K + kc * XK + c0);
        _Float16* ldst = &Wlds[r * XK + c0];
#if USE_ASYNC_LDS
#pragma unroll
        for (int j = 0; j < 4; ++j)
          __builtin_amdgcn_global_load_async_to_lds_b128(
              (gp1_t)(gsrc + j * 8), (lp3_t)(ldst + j * 8), 0, 0);
#else
        const uint4* g4 = (const uint4*)gsrc;
        uint4* l4 = (uint4*)ldst;
        l4[0] = g4[0]; l4[1] = g4[1]; l4[2] = g4[2]; l4[3] = g4[3];
#endif
      }
      // ---- stage activation tile transposed: Xlds[n][k] = src[kc*128+k][l0+n]
      if (mh == 0 || KT > 1) {
        const int n  = tid & 127;
        const int k0 = tid >> 7;        // 0..1
        const int l  = l0 + n;
        for (int k = k0; k < XK; k += 2) {
          float xv = 0.f;
          if (l < LLEN) {
            long off = ((long)s * K + (kc * XK + k)) * LLEN + l;
            xv = SRCF16 ? (float)srcH[off] : srcF[off];
            if constexpr (KT > 1) {
              if (kc + 1 < KT)
                __builtin_prefetch(SRCF16 ? (const void*)&srcH[off + (long)XK * LLEN]
                                          : (const void*)&srcF[off + (long)XK * LLEN], 0, 1);
            }
          }
          Xlds[n * XS + k] = (_Float16)xv;
        }
      }
#if USE_ASYNC_LDS
      __builtin_amdgcn_s_wait_asynccnt(0);
#endif
      __syncthreads();

      // ---- 4 x (16x16x32) WMMA steps across the 128-K chunk.
      // Issue all operand loads first so waits are incremental and next-step
      // LDS loads overlap WMMA execution.
#pragma unroll
      for (int kk = 0; kk < 4; ++kk) {
        const int kkB = kk * 32;
        // B operand: lane (0..15)->col n of col-tile, K-major contiguous
        const _Float16* xp = &Xlds[(wave * 16 + (lane & 15)) * XS + kkB + ((lane >> 4) * 16)];
        h8 bl = *(const h8*)xp;
        h8 bh = *(const h8*)(xp + 8);
        // A operands: lanes<16 hold K 0-15, lanes>=16 hold K 16-31
        h8 al[4], ah[4];
#pragma unroll
        for (int rt = 0; rt < 4; ++rt) {
          const _Float16* wp = &Wlds[(rt * 16 + (lane & 15)) * XK + kkB + ((lane >> 4) * 16)];
          al[rt] = *(const h8*)wp;
          ah[rt] = *(const h8*)(wp + 8);
        }
        h16 bOp;
#pragma unroll
        for (int j = 0; j < 8; ++j) { bOp[j] = bl[j]; bOp[j + 8] = bh[j]; }
#pragma unroll
        for (int rt = 0; rt < 4; ++rt) {
          h16 aOp;
#pragma unroll
          for (int j = 0; j < 8; ++j) { aOp[j] = al[rt][j]; aOp[j + 8] = ah[rt][j]; }
          acc[rt] = __builtin_amdgcn_wmma_f32_16x16x32_f16(
              false, aOp, false, bOp, (short)0, acc[rt], false, false);
        }
      }
      __syncthreads();
    }

    // ---- epilogue: D layout lane<16 -> m=r, lane>=16 -> m=8+r ; n=lane&15
    const int n   = lane & 15;
    const int col = l0 + wave * 16 + n;
    if (col < LLEN) {
#pragma unroll
      for (int rt = 0; rt < 4; ++rt) {
#pragma unroll
        for (int r = 0; r < 8; ++r) {
          const int mrow = mh * 64 + rt * 16 + r + 8 * (lane >> 4);
          const long off = ((long)s * M + mrow) * LLEN + col;
          float a = acc[rt][r];
          float y;
          if (EPI == EPI_SIG) {
            y = 1.f / (1.f + __expf(-a));
          } else {
            float bias = u[mrow] - mean_rstd * v[mrow];
            y = rstd * a + bias;
            if (EPI == EPI_RES) y += dst[off];
          }
          dst[off] = y;
        }
      }
    }
  }
}

// ---------------------------------------------------------------------------
extern "C" void kernel_launch(void* const* d_in, const int* in_sizes, int n_in,
                              void* d_out, int out_size, void* d_ws, size_t ws_size,
                              hipStream_t stream) {
  const float* x      = (const float*)d_in[0];
  const float* gn_g   = (const float*)d_in[1];
  const float* gn_b   = (const float*)d_in[2];
  const float* w_comp = (const float*)d_in[3];
  const float* dw_w   = (const float*)d_in[4];
  const float* pr_a   = (const float*)d_in[5];
  const float* tn_g   = (const float*)d_in[6];
  const float* tn_b   = (const float*)d_in[7];
  const float* pw_w   = (const float*)d_in[8];
  const float* w_exp  = (const float*)d_in[9];
  float* out = (float*)d_out;

  // workspace bump allocator (needs ~26 MB)
  char* p = (char*)d_ws;
  auto carve = [&](size_t bytes) -> void* {
    void* r = (void*)p;
    p += (bytes + 255) & ~size_t(255);
    return r;
  };
  float*    outbuf = (float*)   carve((size_t)SAMPLES * B_CH * LLEN * 4);
  _Float16* tbuf   = (_Float16*)carve((size_t)SAMPLES * B_CH * LLEN * 2);
  _Float16* Wg     = (_Float16*)carve((size_t)B_CH * N_CH * 2);
  _Float16* PWg    = (_Float16*)carve((size_t)NBLK * B_CH * B_CH * 2);
  _Float16* Wexp   = (_Float16*)carve((size_t)NSPK * N_CH * B_CH * 2);
  float*    uvec   = (float*)   carve((size_t)(1 + NBLK) * B_CH * 4);
  float*    vvec   = (float*)   carve((size_t)(1 + NBLK) * B_CH * 4);
  float*    stats  = (float*)   carve((size_t)SAMPLES * 2 * 4);
  float*    parts  = (float*)   carve((size_t)SAMPLES * (B_CH * 32) * 2 * 4);

  const int totW = B_CH * N_CH + NBLK * B_CH * B_CH + NSPK * N_CH * B_CH;
  prep_weights_kernel<<<(totW + 255) / 256, 256, 0, stream>>>(
      w_comp, gn_g, pw_w, tn_g, w_exp, Wg, PWg, Wexp);
  prep_uv_kernel<<<((1 + NBLK) * B_CH + 255) / 256, 256, 0, stream>>>(
      w_comp, gn_g, gn_b, pw_w, tn_g, tn_b, uvec, vvec);

  // stage 0: outer GroupNorm folded into compression GEMM
  reduce_x_kernel<<<dim3(128, SAMPLES), 256, 0, stream>>>(x, parts);
  finalize_stats_kernel<<<SAMPLES, 256, 0, stream>>>(parts, 128, stats);
  gemm_kernel<B_CH, N_CH, EPI_BIAS, false>
      <<<dim3((LLEN + 127) / 128, SAMPLES), 256, 0, stream>>>(
          Wg, x, outbuf, stats, uvec, vvec, 1.f / ((float)N_CH * LLEN));

  // 24 TCN blocks (sequential)
  for (int i = 0; i < NBLK; ++i) {
    const int d = 1 << (i & 7);
    dwprelu_kernel<<<dim3(32, B_CH, SAMPLES), 256, 0, stream>>>(
        outbuf, dw_w, pr_a, tbuf, parts, i, d);
    finalize_stats_kernel<<<SAMPLES, 256, 0, stream>>>(parts, B_CH * 32, stats);
    gemm_kernel<B_CH, B_CH, EPI_RES, true>
        <<<dim3((LLEN + 127) / 128, SAMPLES), 256, 0, stream>>>(
            PWg + (size_t)i * B_CH * B_CH, tbuf, outbuf, stats,
            uvec + (1 + i) * B_CH, vvec + (1 + i) * B_CH,
            1.f / ((float)B_CH * LLEN));
  }

  // expansion + sigmoid -> d_out [4][2*512][8000]
  gemm_kernel<NSPK * N_CH, B_CH, EPI_SIG, false>
      <<<dim3((LLEN + 127) / 128, SAMPLES), 256, 0, stream>>>(
          Wexp, outbuf, out, nullptr, nullptr, nullptr, 0.f);
}
